// RelationalGraphConvLayer_41824391528458
// MI455X (gfx1250) — compile-verified, matrix operand order checked by reading, not run on hardware
//
#include <hip/hip_runtime.h>
#include <hip/hip_bf16.h>
#include <stddef.h>

// ---------------------------------------------------------------------------
// RGCN layer for MI455X (gfx1250, wave32, WMMA).
//   Y_b = X @ w_bases[b]            (WMMA f16 GEMM, f32 accumulate)
//   out[dst] += val * sum_b w_rel[rel,b] * Y_b[src]   (gather + f32 atomics)
// Y stored f16 (102 MB) so the edge-gather phase runs largely out of the
// 192 MB L2 at far above HBM bandwidth.
// GEMM: W transposed+converted once (128 KB) so both A and B fragments load
// as contiguous b128 clauses; B fragments held in VGPRs across 10 m-tiles;
// all loop-invariant addressing hoisted, stores use immediate offsets.
// ---------------------------------------------------------------------------

typedef _Float16 f16;
typedef __attribute__((ext_vector_type(16))) _Float16 v16h;
typedef __attribute__((ext_vector_type(4)))  _Float16 v4h;
typedef __attribute__((ext_vector_type(4)))  float    v4f;
typedef __attribute__((ext_vector_type(8)))  float    v8f;

#define NN      100000
#define EE      640000
#define DD      128
#define RR      8
#define BB      4
#define TILES_M (NN / 16)          /* 6250 */
#define TILES_O (DD / 16)          /* 8    */
#define MSTEPS  10                 /* m-tiles per wave */
#define MGROUPS (TILES_M / MSTEPS) /* 625  */

// ---------------- elementwise helpers ----------------
__global__ void __launch_bounds__(256) cvt_f32_to_f16_kernel(const float* __restrict__ in,
                                                             f16* __restrict__ out, int n) {
  int i = blockIdx.x * 256 + threadIdx.x;
  if (i < n) out[i] = (f16)in[i];
}

// w_bases [B][I][O] f32  ->  Wt [B][O][I] f16  (transpose so B-fragments are
// contiguous-row loads, same shape as A-fragments)
__global__ void __launch_bounds__(256) cvt_transpose_w_kernel(const float* __restrict__ in,
                                                              f16* __restrict__ out) {
  int j = blockIdx.x * 256 + threadIdx.x;      // over B*128*128 = 65536
  int b = j >> 14;
  int i = (j >> 7) & (DD - 1);
  int o = j & (DD - 1);
  out[(size_t)b * (DD * DD) + (size_t)o * DD + i] = (f16)in[j];
}

__global__ void __launch_bounds__(256) zero_f32x4_kernel(v4f* __restrict__ p, int n4) {
  int i = blockIdx.x * 256 + threadIdx.x;
  if (i < n4) p[i] = (v4f){0.f, 0.f, 0.f, 0.f};
}

// ---------------- WMMA basis GEMM: Y[b] = Xh @ W[b] ----------------
// One wave32 per (base, o-tile, m-group); loops MSTEPS m-tiles with B
// fragments resident in VGPRs. K=128 = 4 x (K=32) WMMA steps.
// Fragment packing per CDNA5 ISA 7.12.2 (wave32):
//   A (16x32 f16): lane L holds row M=L&15; VGPR v holds K-pair
//     k = 2*(v&3) + 16*(v>>2) + 8*(L>>4)
//   B (32x16 f16): lane L holds col N=L&15; same K mapping (read from Wt row).
//   C/D (16x16 f32): VGPR v -> row M = v + 8*(L>>4), col N = L&15.
__global__ void __launch_bounds__(256) rgcn_basis_gemm_kernel(const f16* __restrict__ Xh,
                                                              const f16* __restrict__ Wt,
                                                              f16* __restrict__ Y) {
  const int wid  = blockIdx.x * 8 + (threadIdx.x >> 5);   // 8 waves / block
  const int lane = threadIdx.x & 31;
  const int b    = wid / (TILES_O * MGROUPS);
  const int rem  = wid - b * (TILES_O * MGROUPS);
  const int o0   = (rem / MGROUPS) * 16;
  const int mg   = rem % MGROUPS;
  const int half = lane >> 4;              // 0 or 1
  const int lr   = lane & 15;

  // B fragments: one per K-chunk, loaded once, contiguous row of Wt.
  const f16* __restrict__ wrow = Wt + (size_t)b * (DD * DD) + (size_t)(o0 + lr) * DD;
  v16h bfrag[4];
#pragma unroll
  for (int kc = 0; kc < 4; ++kc) {
#pragma unroll
    for (int v = 0; v < 8; ++v) {
      const int k = kc * 32 + 2 * (v & 3) + 16 * (v >> 2) + 8 * half;
      bfrag[kc][2 * v]     = wrow[k];
      bfrag[kc][2 * v + 1] = wrow[k + 1];
    }
  }

  // Loop-invariant bases; per-tile we only bump two pointers by 16 rows.
  const f16* __restrict__ xptr =
      Xh + (size_t)(mg * MSTEPS * 16 + lr) * DD;
  f16* __restrict__ yptr =
      Y + (size_t)b * ((size_t)NN * DD)
        + (size_t)(mg * MSTEPS * 16) * DD
        + (size_t)half * (8 * DD)
        + o0 + lr;

  for (int s = 0; s < MSTEPS; ++s) {
    v8f acc = {};
#pragma unroll
    for (int kc = 0; kc < 4; ++kc) {
      v16h afrag;
#pragma unroll
      for (int v = 0; v < 8; ++v) {
        const int k = kc * 32 + 2 * (v & 3) + 16 * (v >> 2) + 8 * half;
        afrag[2 * v]     = xptr[k];
        afrag[2 * v + 1] = xptr[k + 1];
      }
      acc = __builtin_amdgcn_wmma_f32_16x16x32_f16(
          /*neg_a=*/false, afrag, /*neg_b=*/false, bfrag[kc],
          /*c_mod=*/(short)0, acc, /*reuse_a=*/false, /*reuse_b=*/false);
    }

    // 8 stores, all immediate offsets (v*DD*2 bytes <= 1792).
#pragma unroll
    for (int v = 0; v < 8; ++v) {
      yptr[v * DD] = (f16)acc[v];
    }

    xptr += (size_t)16 * DD;
    yptr += (size_t)16 * DD;
  }
}

// ---------------- edge scatter: out[dst] += val * sum_b c_b * Y_b[src] ------
// One wave32 per edge; each lane covers 4 contiguous output dims (b64 loads
// from each of the 4 basis tables, 4 global_atomic_add_f32 per lane).
__global__ void __launch_bounds__(256) rgcn_edge_scatter_kernel(
    const int* __restrict__ esrc, const int* __restrict__ edst,
    const int* __restrict__ erel, const float* __restrict__ eval,
    const float* __restrict__ w_rel,      // [R,B]
    const f16* __restrict__ Y,            // [B,N,128] f16
    float* __restrict__ out) {            // [N,128] f32
  const int e = blockIdx.x * 8 + (threadIdx.x >> 5);
  if (e >= EE) return;
  const int lane = threadIdx.x & 31;

  const int   src = esrc[e];
  const int   dst = edst[e];
  const int   rel = erel[e];
  const float val = eval[e];

  const float c0 = w_rel[rel * BB + 0] * val;
  const float c1 = w_rel[rel * BB + 1] * val;
  const float c2 = w_rel[rel * BB + 2] * val;
  const float c3 = w_rel[rel * BB + 3] * val;

  const size_t stride = (size_t)NN * DD;
  const size_t base   = (size_t)src * DD + (size_t)lane * 4;
  const v4h y0 = *(const v4h*)(Y + base);
  const v4h y1 = *(const v4h*)(Y + stride + base);
  const v4h y2 = *(const v4h*)(Y + 2 * stride + base);
  const v4h y3 = *(const v4h*)(Y + 3 * stride + base);

  float* __restrict__ o = out + (size_t)dst * DD + (size_t)lane * 4;
#pragma unroll
  for (int j = 0; j < 4; ++j) {
    const float m = c0 * (float)y0[j] + c1 * (float)y1[j] +
                    c2 * (float)y2[j] + c3 * (float)y3[j];
    atomicAdd(o + j, m);
  }
}

// ---------------------------------------------------------------------------
extern "C" void kernel_launch(void* const* d_in, const int* in_sizes, int n_in,
                              void* d_out, int out_size, void* d_ws, size_t ws_size,
                              hipStream_t stream) {
  const float* X       = (const float*)d_in[0];  // [N,128]
  const int*   esrc    = (const int*)  d_in[1];  // [E]
  const int*   edst    = (const int*)  d_in[2];  // [E]
  const int*   erel    = (const int*)  d_in[3];  // [E]
  const float* eval    = (const float*)d_in[4];  // [E]
  const float* w_bases = (const float*)d_in[5];  // [B,128,128]
  const float* w_rel   = (const float*)d_in[6];  // [R,B]
  float*       out     = (float*)d_out;          // [N,128]

  // Workspace layout (f16): Y [B,N,128] | Xh [N,128] | Wt [B,128,128]
  f16* Y  = (f16*)d_ws;
  f16* Xh = Y  + (size_t)BB * NN * DD;
  f16* Wt = Xh + (size_t)NN * DD;

  const int nX = NN * DD;            // 12.8M
  const int nW = BB * DD * DD;       // 65536
  cvt_f32_to_f16_kernel<<<(nX + 255) / 256, 256, 0, stream>>>(X, Xh, nX);
  cvt_transpose_w_kernel<<<nW / 256, 256, 0, stream>>>(w_bases, Wt);
  zero_f32x4_kernel<<<(nX / 4 + 255) / 256, 256, 0, stream>>>((v4f*)out, nX / 4);

  const int gemm_waves = BB * TILES_O * MGROUPS;   // 20000 waves
  rgcn_basis_gemm_kernel<<<gemm_waves / 8, 256, 0, stream>>>(Xh, Wt, Y);

  rgcn_edge_scatter_kernel<<<(EE + 7) / 8, 256, 0, stream>>>(
      esrc, edst, erel, eval, w_rel, Y, out);
}
